// MRAN_60653528154386
// MI455X (gfx1250) — matrix operand order repaired and checked by reading, not compiled, not run on hardware
//
#include <hip/hip_runtime.h>
#include <hip/hip_bf16.h>

#define EPS 1e-5f

typedef __attribute__((ext_vector_type(16))) __bf16 bf16x16;
typedef __attribute__((ext_vector_type(8)))  float  f32x8;

__device__ __forceinline__ unsigned short f2bf(float f) {
  unsigned int u = __float_as_uint(f);
  u += 0x7FFFu + ((u >> 16) & 1u);          // round-to-nearest-even
  return (unsigned short)(u >> 16);
}

// ---------------------------------------------------------------------------
// f32 -> bf16 bulk conversion (n must be a multiple of 8; holds for E, fc_w)
// ---------------------------------------------------------------------------
__global__ __launch_bounds__(256) void f32_to_bf16(const float* __restrict__ src,
                                                   unsigned short* __restrict__ dst,
                                                   int n)
{
  const int stride = gridDim.x * 256 * 8;
  for (int i = (blockIdx.x * 256 + threadIdx.x) * 8; i < n; i += stride) {
    const float4 a = *reinterpret_cast<const float4*>(src + i);
    const float4 b = *reinterpret_cast<const float4*>(src + i + 4);
    uint4 p;
    p.x = (unsigned)f2bf(a.x) | ((unsigned)f2bf(a.y) << 16);
    p.y = (unsigned)f2bf(a.z) | ((unsigned)f2bf(a.w) << 16);
    p.z = (unsigned)f2bf(b.x) | ((unsigned)f2bf(b.y) << 16);
    p.w = (unsigned)f2bf(b.z) | ((unsigned)f2bf(b.w) << 16);
    *reinterpret_cast<uint4*>(dst + i) = p;
  }
}

// ---------------------------------------------------------------------------
// Front kernel: gather + interaction + bn0 + per-sample 1x9 conv + ECA + bn1
// One block per batch element. Produces xo (bf16) [B, 32*2*392 = 25088].
// ---------------------------------------------------------------------------
__global__ __launch_bounds__(256) void mran_front(
    const int* __restrict__ eid, const int* __restrict__ rid,
    const float* __restrict__ E, const float* __restrict__ R,
    const float* __restrict__ Filt, const float* __restrict__ eca_w,
    const float* __restrict__ bn0g, const float* __restrict__ bn0b,
    const float* __restrict__ bn0m, const float* __restrict__ bn0v,
    const float* __restrict__ bn1g, const float* __restrict__ bn1b,
    const float* __restrict__ bn1m, const float* __restrict__ bn1v,
    unsigned short* __restrict__ xo)
{
  constexpr int D = 400, CO = 32, KW = 9, WO = 392;
  __shared__ float Xs[2][D + 8];
  __shared__ float Fs[CO * KW];
  __shared__ float Tsum[2 * KW];
  __shared__ float ya[CO], sS[CO], tS[CO];

  const int b = blockIdx.x;
  const int t = threadIdx.x;
  const int e = eid[b], r = rid[b];

  const float s0 = bn0g[0] * rsqrtf(bn0v[0] + EPS);
  const float c0 = bn0b[0] - bn0m[0] * s0;

  for (int d = t; d < D; d += 256) {
    float uf = E[(size_t)e * D + d];
    float yf = R[(size_t)r * D + d];
    float p  = 2.f * uf * yf;
    Xs[0][d] = (uf + p) * s0 + c0;
    Xs[1][d] = (yf + p) * s0 + c0;
  }
  if (t < 8) { Xs[0][D + t] = 0.f; Xs[1][D + t] = 0.f; }
  for (int i = t; i < CO * KW; i += 256) Fs[i] = Filt[(size_t)r * CO * KW + i];
  if (t < 2 * KW) Tsum[t] = 0.f;
  __syncthreads();

  if (t < 252) {
    int grp = t / 14;                 // 0..17
    int sub = t % 14;
    int hh = grp / KW, k = grp % KW;
    float s = 0.f;
    for (int w = sub; w < WO; w += 14) s += Xs[hh][w + k];
    atomicAdd(&Tsum[grp], s);
  }
  __syncthreads();

  if (t < CO) {
    float s = 0.f;
#pragma unroll
    for (int k = 0; k < KW; ++k) s += Fs[t * KW + k] * (Tsum[k] + Tsum[KW + k]);
    ya[t] = s * (1.f / (2.f * (float)WO));
  }
  __syncthreads();

  if (t < CO) {
    float l = (t > 0)      ? ya[t - 1] : 0.f;
    float c = ya[t];
    float h = (t < CO - 1) ? ya[t + 1] : 0.f;
    float z = l * eca_w[0] + c * eca_w[1] + h * eca_w[2];
    float att = 1.f / (1.f + __expf(-z));
    float rs = bn1g[t] * rsqrtf(bn1v[t] + EPS);
    sS[t] = att * rs;
    tS[t] = bn1b[t] - bn1m[t] * rs;
  }
  __syncthreads();

  unsigned short* out = xo + (size_t)b * (CO * 2 * WO);
  for (int idx = t; idx < CO * 2 * WO; idx += 256) {
    int o   = idx / (2 * WO);
    int rem = idx - o * (2 * WO);
    int hh  = rem / WO;
    int w   = rem - hh * WO;
    float acc = 0.f;
#pragma unroll
    for (int k = 0; k < KW; ++k) acc += Xs[hh][w + k] * Fs[o * KW + k];
    float v = fmaxf(acc * sS[o] + tS[o], 0.f);
    out[idx] = f2bf(v);
  }
}

// ---------------------------------------------------------------------------
// Tiled bf16 WMMA GEMM:  C[M,N] = A[M,K](bf16) * B[N,K](bf16, used as K x N)
// 256 threads = 8 waves (2x4); block tile 64x128, K-step 32; wave tile 32x32.
// Staging uses gfx1250 async global->LDS DMA (ASYNCcnt) with double-buffered
// LDS: tile k+1 copies overlap tile k WMMA; one barrier per iteration.
// K%16==0 makes every staged segment all-or-nothing (zero ds_store fallback).
// EPI==0: relu(bn2(acc + bias)) -> bf16     EPI==1: sigmoid(acc + bias) -> f32
// ---------------------------------------------------------------------------

// Stage one 64x32 A-segment + one 128x32 B-segment of tile starting at K0N
// into LDS buffer BUF using async global->LDS b128 copies.
#define STAGE_TILES(K0N, BUF)                                                   \
  {                                                                             \
    const int kkA = (K0N) + aseg * 8;                                           \
    if (kkA + 8 <= K) {                                                         \
      unsigned ldsA =                                                           \
          (unsigned)(size_t)&As[(BUF)][arow * LD + aseg * 8];                   \
      unsigned long long gaA =                                                  \
          (unsigned long long)(size_t)&A[(size_t)(m0 + arow) * K + kkA];        \
      asm volatile("global_load_async_to_lds_b128 %0, %1, off"                  \
                   :: "v"(ldsA), "v"(gaA) : "memory");                          \
    } else {                                                                    \
      uint4 z = {0u, 0u, 0u, 0u};                                               \
      *reinterpret_cast<uint4*>(&As[(BUF)][arow * LD + aseg * 8]) = z;          \
    }                                                                           \
    const int kkB = (K0N) + bseg * 16;                                          \
    if (kkB + 16 <= K) {                                                        \
      unsigned ldsB =                                                           \
          (unsigned)(size_t)&Bs[(BUF)][brow * LD + bseg * 16];                  \
      unsigned long long gaB =                                                  \
          (unsigned long long)(size_t)&B[(size_t)gbn * K + kkB];                \
      asm volatile("global_load_async_to_lds_b128 %0, %1, off\n\t"              \
                   "global_load_async_to_lds_b128 %0, %1, off offset:16"        \
                   :: "v"(ldsB), "v"(gaB) : "memory");                          \
    } else {                                                                    \
      uint4 z = {0u, 0u, 0u, 0u};                                               \
      uint4* q = reinterpret_cast<uint4*>(&Bs[(BUF)][brow * LD + bseg * 16]);   \
      q[0] = z; q[1] = z;                                                       \
    }                                                                           \
  }

template <int EPI>
__global__ __launch_bounds__(256) void wmma_gemm(
    const unsigned short* __restrict__ A,
    const unsigned short* __restrict__ B,
    void* __restrict__ Out,
    const float* __restrict__ bias,
    const float* __restrict__ g2, const float* __restrict__ b2,
    const float* __restrict__ m2, const float* __restrict__ v2,
    int M, int N, int K)
{
  constexpr int BM = 64, BN = 128, BK = 32, LD = 40;  // LD*2 = 80B, 16B aligned
  __shared__ unsigned short As[2][BM * LD];
  __shared__ unsigned short Bs[2][BN * LD];

  const int tid  = threadIdx.x;
  const int lane = tid & 31;
  const int wave = tid >> 5;
  const int wr = wave >> 2, wc = wave & 3;            // 2x4 wave grid
  const int m0 = blockIdx.y * BM;
  const int n0 = blockIdx.x * BN;

  const int arow = tid >> 2, aseg = tid & 3;          // A: 64 rows x 4 segs of 8
  const int brow = tid >> 1, bseg = tid & 1;          // B: 128 rows x 2 segs of 16
  const int gbn  = min(n0 + brow, N - 1);             // branch-free N clamp

  const int fm = lane & 15, fh = lane >> 4;

  f32x8 acc00 = {}, acc01 = {}, acc10 = {}, acc11 = {};

  // Prologue: stage first tile into buffer 0
  STAGE_TILES(0, 0);
  asm volatile("s_wait_asynccnt 0x0" ::: "memory");
  __syncthreads();

  int cur = 0;
#pragma unroll 1
  for (int k0 = 0; k0 < K; k0 += BK) {
    // Kick off async copy of the NEXT tile into the other buffer (overlaps WMMA)
    if (k0 + BK < K) STAGE_TILES(k0 + BK, cur ^ 1);

    // ---- fragments per documented wave32 layouts; 4 WMMAs / wave / step ----
    {
      union { bf16x16 v; unsigned int u[8]; } a0, a1, b0, b1;
      const unsigned short* ar0 = &As[cur][(wr * 32 + fm) * LD];
      const unsigned short* ar1 = &As[cur][(wr * 32 + 16 + fm) * LD];
      const unsigned short* br0 = &Bs[cur][(wc * 32 + fm) * LD];
      const unsigned short* br1 = &Bs[cur][(wc * 32 + 16 + fm) * LD];
#pragma unroll
      for (int i = 0; i < 4; ++i) {
        a0.u[i]     = *reinterpret_cast<const unsigned int*>(&ar0[fh * 8 + 2 * i]);
        a0.u[4 + i] = *reinterpret_cast<const unsigned int*>(&ar0[16 + fh * 8 + 2 * i]);
        a1.u[i]     = *reinterpret_cast<const unsigned int*>(&ar1[fh * 8 + 2 * i]);
        a1.u[4 + i] = *reinterpret_cast<const unsigned int*>(&ar1[16 + fh * 8 + 2 * i]);
      }
#pragma unroll
      for (int i = 0; i < 8; ++i) {
        b0.u[i] = *reinterpret_cast<const unsigned int*>(&br0[fh * 8 + 2 * i + (i >> 2) * 8]);
        b1.u[i] = *reinterpret_cast<const unsigned int*>(&br1[fh * 8 + 2 * i + (i >> 2) * 8]);
      }
      acc00 = __builtin_amdgcn_wmma_f32_16x16x32_bf16(false, a0.v, false, b0.v,
                                                      (short)0, acc00, false, false);
      acc01 = __builtin_amdgcn_wmma_f32_16x16x32_bf16(false, a0.v, false, b1.v,
                                                      (short)0, acc01, false, false);
      acc11 = __builtin_amdgcn_wmma_f32_16x16x32_bf16(false, a1.v, false, b1.v,
                                                      (short)0, acc11, false, false);
      acc10 = __builtin_amdgcn_wmma_f32_16x16x32_bf16(false, a1.v, false, b0.v,
                                                      (short)0, acc10, false, false);
    }

    // Join own async copies, then barrier (its dscnt wait also proves all
    // waves finished reading the buffer the next iteration overwrites).
    asm volatile("s_wait_asynccnt 0x0" ::: "memory");
    __syncthreads();
    cur ^= 1;
  }

  // ---- fused epilogue ----
#pragma unroll
  for (int ni = 0; ni < 2; ++ni) {
    const int gn = n0 + wc * 32 + ni * 16 + fm;
    if (gn >= N) continue;
    const f32x8 ca = (ni == 0) ? acc00 : acc01;
    const f32x8 cb = (ni == 0) ? acc10 : acc11;
    const float bn = bias[gn];
    if (EPI == 0) {
      const float rs = g2[gn] * rsqrtf(v2[gn] + EPS);
      const float tt = b2[gn] - m2[gn] * rs;
      unsigned short* O = reinterpret_cast<unsigned short*>(Out);
#pragma unroll
      for (int rr = 0; rr < 8; ++rr) {
        const int gm = m0 + wr * 32 + fh * 8 + rr;
        O[(size_t)gm * N + gn]        = f2bf(fmaxf((ca[rr] + bn) * rs + tt, 0.f));
        O[(size_t)(gm + 16) * N + gn] = f2bf(fmaxf((cb[rr] + bn) * rs + tt, 0.f));
      }
    } else {
      float* O = reinterpret_cast<float*>(Out);
#pragma unroll
      for (int rr = 0; rr < 8; ++rr) {
        const int gm = m0 + wr * 32 + fh * 8 + rr;
        O[(size_t)gm * N + gn]        = 1.f / (1.f + __expf(-(ca[rr] + bn)));
        O[(size_t)(gm + 16) * N + gn] = 1.f / (1.f + __expf(-(cb[rr] + bn)));
      }
    }
  }
}

// ---------------------------------------------------------------------------
extern "C" void kernel_launch(void* const* d_in, const int* in_sizes, int n_in,
                              void* d_out, int out_size, void* d_ws, size_t ws_size,
                              hipStream_t stream)
{
  const int*   entity_id   = (const int*)d_in[0];
  const int*   relation_id = (const int*)d_in[1];
  const float* E      = (const float*)d_in[2];
  const float* R      = (const float*)d_in[3];
  const float* Filt   = (const float*)d_in[4];
  const float* fc_w   = (const float*)d_in[5];
  const float* fc_b   = (const float*)d_in[6];
  const float* eca_w  = (const float*)d_in[7];
  const float* out_bias = (const float*)d_in[8];
  const float* bn0g = (const float*)d_in[9],  *bn0b = (const float*)d_in[10];
  const float* bn0m = (const float*)d_in[11], *bn0v = (const float*)d_in[12];
  const float* bn1g = (const float*)d_in[13], *bn1b = (const float*)d_in[14];
  const float* bn1m = (const float*)d_in[15], *bn1v = (const float*)d_in[16];
  const float* bn2g = (const float*)d_in[17], *bn2b = (const float*)d_in[18];
  const float* bn2m = (const float*)d_in[19], *bn2v = (const float*)d_in[20];

  const int B = 1024, D = 400, ENT = 40943;
  const int KFC = 32 * 2 * 392;            // 25088

  // Workspace layout (all 16B aligned)
  unsigned short* xo  = (unsigned short*)d_ws;                 // [1024, 25088] bf16
  unsigned short* h   = xo  + (size_t)B * KFC;                 // [1024, 400]   bf16
  unsigned short* Ebf = h   + (size_t)B * D;                   // [40943, 400]  bf16
  unsigned short* Wbf = Ebf + (size_t)ENT * D;                 // [400, 25088]  bf16

  // One-time (per launch) f32 -> bf16 conversions; both stay L2-resident.
  f32_to_bf16<<<2048, 256, 0, stream>>>(E,    Ebf, ENT * D);
  f32_to_bf16<<<2048, 256, 0, stream>>>(fc_w, Wbf, D * KFC);

  mran_front<<<B, 256, 0, stream>>>(entity_id, relation_id, E, R, Filt, eca_w,
                                    bn0g, bn0b, bn0m, bn0v,
                                    bn1g, bn1b, bn1m, bn1v, xo);

  // h = relu(bn2(xo @ fc_w^T + fc_b))   [1024 x 400]
  {
    dim3 grid((D + 127) / 128, B / 64);
    wmma_gemm<0><<<grid, 256, 0, stream>>>(xo, Wbf, (void*)h, fc_b,
                                           bn2g, bn2b, bn2m, bn2v, B, D, KFC);
  }
  // out = sigmoid(h @ E^T + out_bias)   [1024 x 40943]
  {
    dim3 grid((ENT + 127) / 128, B / 64);
    wmma_gemm<1><<<grid, 256, 0, stream>>>(h, Ebf, d_out, out_bias,
                                           nullptr, nullptr, nullptr, nullptr,
                                           B, ENT, D);
  }
  (void)in_sizes; (void)n_in; (void)out_size; (void)ws_size;
}
